// GraphMSG_25503515803964
// MI455X (gfx1250) — compile-verified, matrix operand order checked by reading, not run on hardware
//
#include <hip/hip_runtime.h>
#include <hip/hip_bf16.h>

// ---------------- problem constants (AIFS-like GNN, bse == 1) ----------------
constexpr int ERA   = 40320;
constexpr int Hn    = 10242;
constexpr int E_E2H = 100000;
constexpr int E_H2E = 100000;
constexpr int E_H2H = 81920;
constexpr int Cc    = 256;    // hidden channels
constexpr int PCr   = 257;    // processor hidden (real)
constexpr int PCP   = 320;    // processor hidden padded to mult of 64
constexpr float LN_EPS = 1e-5f;

// ---------------- types for WMMA ----------------
typedef __bf16 bf16_t;
typedef __bf16 bf16x8  __attribute__((ext_vector_type(8)));
typedef __bf16 bf16x16 __attribute__((ext_vector_type(16)));
typedef float  f32x8   __attribute__((ext_vector_type(8)));

__device__ __forceinline__ bf16_t f2bf(float x) { return (bf16_t)x; }

__device__ __forceinline__ bf16x16 load_frag16(const bf16_t* base) {
  // 16 bf16 for one WMMA operand lane: two contiguous 8-element (16B) chunks 16 apart.
  bf16x8 lo = *reinterpret_cast<const bf16x8*>(base);
  bf16x8 hi = *reinterpret_cast<const bf16x8*>(base + 16);
  return __builtin_shufflevector(lo, hi, 0,1,2,3,4,5,6,7,8,9,10,11,12,13,14,15);
}

// =====================================================================
// WMMA GEMM:  C[M,Np] = act(A[M,K](bf16) @ Bt[Np,K]^T(bf16) + bias)
// Block tile 256x64, 8 wave32s; wave w owns rows [32w,32w+32) as 2 M-subtiles,
// 4 N-subtiles of 16 => 8x v_wmma_f32_16x16x32_bf16 per 32-deep K step.
// B fragments are hoisted to registers and reused across both M-subtiles.
// =====================================================================
__global__ __launch_bounds__(256) void k_gemm(
    const bf16_t* __restrict__ A, int lda,
    const bf16_t* __restrict__ Bt, int ldb,
    const float* __restrict__ bias, int Nreal,
    int M, int Np, int K,
    float* __restrict__ Cf, int ldc,
    bf16_t* __restrict__ Cb, int ldcb, int act)
{
  __shared__ __align__(16) bf16_t As[256 * 40];   // 256 rows, 32 cols, stride 40 (80B, 16B aligned)
  __shared__ __align__(16) bf16_t Bs[64 * 40];

  const int tid  = threadIdx.x;
  const int lane = tid & 31;
  const int wave = tid >> 5;
  const int mbase = blockIdx.x * 256;
  const int nbase = blockIdx.y * 64;

  f32x8 acc[2][4];
#pragma unroll
  for (int i = 0; i < 2; ++i)
#pragma unroll
    for (int j = 0; j < 4; ++j)
#pragma unroll
      for (int r = 0; r < 8; ++r) acc[i][j][r] = 0.f;

  const int kb = (lane >> 4) * 8;                // per-lane K chunk base (0 or 8)

  for (int k0 = 0; k0 < K; k0 += 32) {
    // stage A tile (256 x 32): 512 half-row pieces, 2 per thread
#pragma unroll
    for (int q = tid; q < 512; q += 256) {
      int r = q >> 1, h = q & 1;
      int gm = mbase + r;
      bf16x8 v0, v1;
#pragma unroll
      for (int i = 0; i < 8; ++i) { v0[i] = (bf16_t)0.f; v1[i] = (bf16_t)0.f; }
      if (gm < M) {
        const bf16_t* pA = A + (size_t)gm * lda + k0 + h * 16;
        v0 = *reinterpret_cast<const bf16x8*>(pA);
        v1 = *reinterpret_cast<const bf16x8*>(pA + 8);
        if (k0 + 32 < K) __builtin_prefetch(pA + 32, 0, 3);   // near-scope global_prefetch_b8
      }
      *reinterpret_cast<bf16x8*>(&As[r * 40 + h * 16])     = v0;
      *reinterpret_cast<bf16x8*>(&As[r * 40 + h * 16 + 8]) = v1;
    }
    if (tid < 128) { // stage Bt tile (64 x 32); Bt padded to Np rows, always valid
      int r = tid >> 1, h = tid & 1;
      const bf16_t* pB = Bt + (size_t)(nbase + r) * ldb + k0 + h * 16;
      *reinterpret_cast<bf16x8*>(&Bs[r * 40 + h * 16])     = *reinterpret_cast<const bf16x8*>(pB);
      *reinterpret_cast<bf16x8*>(&Bs[r * 40 + h * 16 + 8]) = *reinterpret_cast<const bf16x8*>(pB + 8);
    }
    __syncthreads();

    bf16x16 bfrag[4];
#pragma unroll
    for (int j = 0; j < 4; ++j)
      bfrag[j] = load_frag16(&Bs[(j * 16 + (lane & 15)) * 40 + kb]);
#pragma unroll
    for (int i = 0; i < 2; ++i) {
      bf16x16 af = load_frag16(&As[(wave * 32 + i * 16 + (lane & 15)) * 40 + kb]);
#pragma unroll
      for (int j = 0; j < 4; ++j)
        acc[i][j] = __builtin_amdgcn_wmma_f32_16x16x32_bf16(
            false, af, false, bfrag[j], (short)0, acc[i][j], false, false);
    }
    __syncthreads();
  }

  // epilogue: C/D layout — lanes 0-15: N=lane, M=r ; lanes 16-31: N=lane-16, M=8+r
  const int nloc = lane & 15;
  const int mloc = (lane >> 4) * 8;
#pragma unroll
  for (int i = 0; i < 2; ++i) {
#pragma unroll
    for (int j = 0; j < 4; ++j) {
      int n = nbase + j * 16 + nloc;
      float bv = (bias != nullptr && n < Nreal) ? bias[n] : 0.f;
#pragma unroll
      for (int r = 0; r < 8; ++r) {
        int m = mbase + wave * 32 + i * 16 + mloc + r;
        if (m < M) {
          float v = acc[i][j][r] + bv;
          if (act) v = v / (1.f + __expf(-v));    // SiLU
          if (Cf) Cf[(size_t)m * ldc + n]  = v;
          if (Cb) Cb[(size_t)m * ldcb + n] = f2bf(v);
        }
      }
    }
  }
}

// ---------------- LayerNorm (+ optional residual), one wave32 per row ----------------
__global__ __launch_bounds__(256) void k_layernorm(
    const float* __restrict__ X, int ldx, int Nreal, int M,
    const float* __restrict__ g, const float* __restrict__ be,
    const float* __restrict__ resid, int ldr,
    float* __restrict__ Yf, bf16_t* __restrict__ Yb, int ldy)
{
  int lane = threadIdx.x & 31;
  int row  = blockIdx.x * 8 + (threadIdx.x >> 5);
  if (row >= M) return;
  const float* xr = X + (size_t)row * ldx;
  float s1 = 0.f, s2 = 0.f;
  for (int c = lane; c < Nreal; c += 32) { float v = xr[c]; s1 += v; s2 += v * v; }
  for (int o = 16; o > 0; o >>= 1) { s1 += __shfl_xor(s1, o, 32); s2 += __shfl_xor(s2, o, 32); }
  float inv  = 1.f / (float)Nreal;
  float mu   = s1 * inv;
  float var  = fmaxf(s2 * inv - mu * mu, 0.f);
  float rstd = rsqrtf(var + LN_EPS);
  for (int c = lane; c < ldy; c += 32) {
    float y = 0.f;
    if (c < Nreal) {
      y = (xr[c] - mu) * rstd * g[c] + be[c];
      if (resid) y += resid[(size_t)row * ldr + c];
    }
    if (Yf) Yf[(size_t)row * ldy + c] = y;
    if (Yb) Yb[(size_t)row * ldy + c] = f2bf(y);
  }
}

// ---------------- segment_sum via fp32 global atomics ----------------
__global__ void k_scatter_add(const float* __restrict__ m, int ldm,
                              const int* __restrict__ dsti, long long E, int h,
                              float* __restrict__ agg)
{
  long long idx = (long long)blockIdx.x * blockDim.x + threadIdx.x;
  if (idx >= E * (long long)h) return;
  int e = (int)(idx / h), c = (int)(idx % h);
  unsafeAtomicAdd(&agg[(size_t)dsti[e] * ldm + c], m[(size_t)e * ldm + c]);
}

__global__ void k_zero(float* __restrict__ p, long long n) {
  long long i = (long long)blockIdx.x * blockDim.x + threadIdx.x;
  if (i < n) p[i] = 0.f;
}

// ---------------- gather/concat builders (fp32 -> padded bf16 rows) ----------------
__global__ void k_edge_concat(const float* __restrict__ xs, int lds,
                              const float* __restrict__ xd, int ldd,
                              const float* __restrict__ ea, int lde,
                              const int* __restrict__ src, const int* __restrict__ dsti,
                              int h, long long E, bf16_t* __restrict__ out, int Kp)
{
  long long idx = (long long)blockIdx.x * blockDim.x + threadIdx.x;
  if (idx >= E * (long long)Kp) return;
  int e = (int)(idx / Kp), k = (int)(idx % Kp);
  float v = 0.f;
  if (k < h)          v = xs[(size_t)src[e] * lds + k];
  else if (k < 2 * h) v = xd[(size_t)dsti[e] * ldd + (k - h)];
  else if (k < 3 * h) v = ea[(size_t)e * lde + (k - 2 * h)];
  out[idx] = f2bf(v);
}

__global__ void k_node_concat(const float* __restrict__ xd, int ldd,
                              const float* __restrict__ agg, int lda_,
                              int h, long long M, bf16_t* __restrict__ out, int Kp)
{
  long long idx = (long long)blockIdx.x * blockDim.x + threadIdx.x;
  if (idx >= M * (long long)Kp) return;
  int m = (int)(idx / Kp), k = (int)(idx % Kp);
  float v = 0.f;
  if (k < h)          v = xd[(size_t)m * ldd + k];
  else if (k < 2 * h) v = agg[(size_t)m * lda_ + (k - h)];
  out[idx] = f2bf(v);
}

__global__ void k_concat_attr(const float* __restrict__ a, int wa,
                              const float* __restrict__ b, int wb,
                              long long M, bf16_t* __restrict__ out, int Kp)
{
  long long idx = (long long)blockIdx.x * blockDim.x + threadIdx.x;
  if (idx >= M * (long long)Kp) return;
  int m = (int)(idx / Kp), k = (int)(idx % Kp);
  float v = 0.f;
  if (k < wa)           v = a[(size_t)m * wa + k];
  else if (k < wa + wb) v = b[(size_t)m * wb + (k - wa)];
  out[idx] = f2bf(v);
}

// x:(1,1,2,ERA,101) -> [x(m=0)|x(m=1)|latlon(4)|tr(8)] padded to 224
__global__ void k_build_xera(const float* __restrict__ x, const float* __restrict__ ll,
                             const float* __restrict__ tr, bf16_t* __restrict__ out)
{
  long long idx = (long long)blockIdx.x * blockDim.x + threadIdx.x;
  if (idx >= (long long)ERA * 224) return;
  int n = (int)(idx / 224), k = (int)(idx % 224);
  float v = 0.f;
  if (k < 101)      v = x[(size_t)n * 101 + k];
  else if (k < 202) v = x[(size_t)(ERA + n) * 101 + (k - 101)];
  else if (k < 206) v = ll[n * 4 + (k - 202)];
  else if (k < 214) v = tr[n * 8 + (k - 206)];
  out[idx] = f2bf(v);
}

// xp = [x_latent(256) | z(1) | zero-pad] as fp32 & bf16, stride PCP
__global__ void k_build_xp(const float* __restrict__ xl, const float* __restrict__ z,
                           float* __restrict__ xf, bf16_t* __restrict__ xb)
{
  long long idx = (long long)blockIdx.x * blockDim.x + threadIdx.x;
  if (idx >= (long long)Hn * PCP) return;
  int n = (int)(idx / PCP), k = (int)(idx % PCP);
  float v = 0.f;
  if (k < Cc) v = xl[(size_t)n * Cc + k];
  else if (k == Cc) v = z[n];
  xf[idx] = v; xb[idx] = f2bf(v);
}

// x_latent_proc = xp[:, :256] + x_latent
__global__ void k_residual_slice(const float* __restrict__ xp, const float* __restrict__ xl,
                                 float* __restrict__ of, bf16_t* __restrict__ ob)
{
  long long idx = (long long)blockIdx.x * blockDim.x + threadIdx.x;
  if (idx >= (long long)Hn * Cc) return;
  int n = (int)(idx / Cc), c = (int)(idx % Cc);
  float v = xp[(size_t)n * PCP + c] + xl[idx];
  of[idx] = v; ob[idx] = f2bf(v);
}

// weights: W (K x N) fp32 -> Wt (Np x Kp) bf16 (transposed, zero-padded)
__global__ void k_wt(const float* __restrict__ W, int K, int N,
                     bf16_t* __restrict__ Wt, int Kp, int Np)
{
  long long idx = (long long)blockIdx.x * blockDim.x + threadIdx.x;
  if (idx >= (long long)Kp * Np) return;
  int n = (int)(idx / Kp), k = (int)(idx % Kp);
  Wt[idx] = (n < N && k < K) ? f2bf(W[(size_t)k * N + n]) : (bf16_t)0.f;
}

// out = y[:, :98] + x[0,0,1,:, :98]
__global__ void k_final_out(const float* __restrict__ y, const float* __restrict__ x,
                            float* __restrict__ out)
{
  long long idx = (long long)blockIdx.x * blockDim.x + threadIdx.x;
  if (idx >= (long long)ERA * 98) return;
  int n = (int)(idx / 98), f = (int)(idx % 98);
  out[idx] = y[(size_t)n * 128 + f] + x[(size_t)(ERA + n) * 101 + f];
}

// =====================================================================
// host orchestration
// =====================================================================
extern "C" void kernel_launch(void* const* d_in, const int* in_sizes, int n_in,
                              void* d_out, int out_size, void* d_ws, size_t ws_size,
                              hipStream_t stream)
{
  (void)in_sizes; (void)n_in; (void)out_size; (void)ws_size;

  // ---- input indices (setup_inputs order; params pytree sorted-key flatten) ----
  enum {
    IN_X = 0, IN_Z, IN_ERALL, IN_HLL, IN_E2H_A, IN_H2H_A, IN_H2E_A,
    IN_E2H_I, IN_H2H_I, IN_H2E_I, P0 = 10,
    PD_BLK_E = P0 + 0,  PD_BLK_N = P0 + 6,  PD_DST = P0 + 12, PD_EDGE = P0 + 18,
    PD_FIN   = P0 + 24, PD_SRC   = P0 + 28, P_E2HTR = P0 + 34,
    PE_BLK_E = P0 + 35, PE_BLK_N = P0 + 41, PE_DST = P0 + 47, PE_EDGE = P0 + 53,
    PE_SRC   = P0 + 59, P_ERATR  = P0 + 65, P_H2ETR = P0 + 66, P_H2HTR = P0 + 67,
    P_HTR    = P0 + 68, PP_B0_E  = P0 + 69, PP_B0_N = P0 + 75,
    PP_B1_E  = P0 + 81, PP_B1_N  = P0 + 87, PP_EDGE = P0 + 93,
  };
  auto pf = [&](int i) -> const float* { return (const float*)d_in[i]; };

  const float* x = pf(IN_X);
  const int* e2h_src = (const int*)d_in[IN_E2H_I]; const int* e2h_dst = e2h_src + E_E2H;
  const int* h2h_src = (const int*)d_in[IN_H2H_I]; const int* h2h_dst = h2h_src + E_H2H;
  const int* h2e_src = (const int*)d_in[IN_H2E_I]; const int* h2e_dst = h2e_src + E_H2E;

  // ---- workspace bump allocator ----
  char* base = (char*)d_ws; size_t off = 0;
  auto alloc = [&](size_t bytes) -> void* {
    void* r = base + off; off = (off + bytes + 255) & ~(size_t)255; return r;
  };
  constexpr size_t SHARED_CAP = 26214400;              // max(E*256, EH*320)
  bf16_t* XERA_B    = (bf16_t*)alloc((size_t)ERA * 224 * 2);
  bf16_t* XH_B      = (bf16_t*)alloc((size_t)Hn * 32 * 2);
  bf16_t* EAIN_B    = (bf16_t*)alloc((size_t)E_E2H * 32 * 2);
  bf16_t* EDGE_IN_B = (bf16_t*)alloc((size_t)E_E2H * 768 * 2);     // >= EH*800
  bf16_t* NODE_IN_B = (bf16_t*)alloc((size_t)ERA * 512 * 2);       // >= Hn*544
  bf16_t* HID_B     = (bf16_t*)alloc(SHARED_CAP * 2);
  float*  GOUT_F    = (float*)alloc(SHARED_CAP * 4);
  float*  EA_F      = (float*)alloc(SHARED_CAP * 4);
  bf16_t* EA_B      = (bf16_t*)alloc(SHARED_CAP * 2);
  float*  AGG_F     = (float*)alloc((size_t)ERA * Cc * 4);
  float*  S_F  = (float*)alloc((size_t)ERA * Cc * 4);  bf16_t* S_B  = (bf16_t*)alloc((size_t)ERA * Cc * 2);
  float*  D_F  = (float*)alloc((size_t)Hn * Cc * 4);   bf16_t* D_B  = (bf16_t*)alloc((size_t)Hn * Cc * 2);
  float*  XP_F = (float*)alloc((size_t)Hn * PCP * 4);  bf16_t* XP_B = (bf16_t*)alloc((size_t)Hn * PCP * 2);
  float*  PR_F = (float*)alloc((size_t)Hn * Cc * 4);   bf16_t* PR_B = (bf16_t*)alloc((size_t)Hn * Cc * 2);
  float*  DS_F = (float*)alloc((size_t)Hn * Cc * 4);   bf16_t* DS_B = (bf16_t*)alloc((size_t)Hn * Cc * 2);
  float*  DD_F = (float*)alloc((size_t)ERA * Cc * 4);  bf16_t* DD_B = (bf16_t*)alloc((size_t)ERA * Cc * 2);

  auto G1 = [](long long n) { return dim3((unsigned)((n + 255) / 256)); };

  // ---- weight conversion (fp32 KxN -> bf16 NpxKp transposed) ----
  auto conv = [&](const float* W, int K, int N, int Kp, int Np) -> bf16_t* {
    bf16_t* Wt = (bf16_t*)alloc((size_t)Kp * Np * 2);
    k_wt<<<G1((long long)Kp * Np), 256, 0, stream>>>(W, K, N, Wt, Kp, Np);
    return Wt;
  };
  struct Mlp { const bf16_t *W0, *W1; const float *b0, *b1, *g, *be; int K0p, Np, N; };
  auto mkmlp = [&](int b, int K0, int K0p, int N, int Np) -> Mlp {
    Mlp m; m.K0p = K0p; m.Np = Np; m.N = N;
    m.W0 = conv(pf(b + 0), K0, N, K0p, Np);
    m.W1 = conv(pf(b + 1), N,  N, Np,  Np);
    m.b0 = pf(b + 2); m.b1 = pf(b + 3); m.g = pf(b + 4); m.be = pf(b + 5);
    return m;
  };

  Mlp encSrc  = mkmlp(PE_SRC,  214, 224, Cc, Cc);
  Mlp encDst  = mkmlp(PE_DST,  12,  32,  Cc, Cc);
  Mlp encEdge = mkmlp(PE_EDGE, 11,  32,  Cc, Cc);
  Mlp encBE   = mkmlp(PE_BLK_E, 768, 768, Cc, Cc);
  Mlp encBN   = mkmlp(PE_BLK_N, 512, 512, Cc, Cc);
  Mlp prEdge  = mkmlp(PP_EDGE, 11,  32,  PCr, PCP);
  Mlp prB0E   = mkmlp(PP_B0_E, 771, 800, PCr, PCP);
  Mlp prB0N   = mkmlp(PP_B0_N, 514, 544, PCr, PCP);
  Mlp prB1E   = mkmlp(PP_B1_E, 771, 800, PCr, PCP);
  Mlp prB1N   = mkmlp(PP_B1_N, 514, 544, PCr, PCP);
  Mlp decSrc  = mkmlp(PD_SRC,  Cc, Cc, Cc, Cc);
  Mlp decDst  = mkmlp(PD_DST,  Cc, Cc, Cc, Cc);
  Mlp decEdge = mkmlp(PD_EDGE, 11, 32, Cc, Cc);
  Mlp decBE   = mkmlp(PD_BLK_E, 768, 768, Cc, Cc);
  Mlp decBN   = mkmlp(PD_BLK_N, 512, 512, Cc, Cc);
  bf16_t* FW0 = conv(pf(PD_FIN + 0), Cc, Cc, Cc, Cc);
  bf16_t* FW1 = conv(pf(PD_FIN + 1), Cc, 98, Cc, 128);

  // ---- launch helpers ----
  auto gemmL = [&](const bf16_t* A, int lda, const bf16_t* Bt, int ldb, const float* bias,
                   int Nreal, int M, int Np, int K, float* Cf, int ldc,
                   bf16_t* Cb, int ldcb, int act) {
    dim3 g((unsigned)((M + 255) / 256), (unsigned)(Np / 64));
    k_gemm<<<g, 256, 0, stream>>>(A, lda, Bt, ldb, bias, Nreal, M, Np, K, Cf, ldc, Cb, ldcb, act);
  };
  auto run_mlp = [&](const Mlp& m, const bf16_t* Xb, int lda, int M,
                     const float* resid, int ldr, float* Yf, bf16_t* Yb, int ldy) {
    gemmL(Xb, lda, m.W0, m.K0p, m.b0, m.N, M, m.Np, m.K0p, nullptr, 0, HID_B, m.Np, 1);
    gemmL(HID_B, m.Np, m.W1, m.Np, m.b1, m.N, M, m.Np, m.Np, GOUT_F, m.Np, nullptr, 0, 1);
    k_layernorm<<<dim3((unsigned)((M + 7) / 8)), 256, 0, stream>>>(
        GOUT_F, m.Np, m.N, M, m.g, m.be, resid, ldr, Yf, Yb, ldy);
  };
  auto run_block = [&](const Mlp& em, const Mlp& nm, const float* xs, int lds,
                       float* xd, bf16_t* xdB, int ldd, float* ea, bf16_t* eaB, int lde,
                       int h, const int* src, const int* dsti, long long E, int n_dst,
                       int eKp, int nKp) {
    k_edge_concat<<<G1(E * eKp), 256, 0, stream>>>(xs, lds, xd, ldd, ea, lde,
                                                   src, dsti, h, E, EDGE_IN_B, eKp);
    run_mlp(em, EDGE_IN_B, eKp, (int)E, ea, lde, ea, eaB, lde);          // m -> ea (+resid)
    long long aggN = (long long)n_dst * lde;
    k_zero<<<G1(aggN), 256, 0, stream>>>(AGG_F, aggN);
    k_scatter_add<<<G1(E * h), 256, 0, stream>>>(ea, lde, dsti, E, h, AGG_F);
    k_node_concat<<<G1((long long)n_dst * nKp), 256, 0, stream>>>(
        xd, ldd, AGG_F, lde, h, (long long)n_dst, NODE_IN_B, nKp);
    run_mlp(nm, NODE_IN_B, nKp, n_dst, xd, ldd, xd, xdB, ldd);           // xd += ...
  };

  // ================= encoder (era -> h) =================
  k_build_xera<<<G1((long long)ERA * 224), 256, 0, stream>>>(x, pf(IN_ERALL), pf(P_ERATR), XERA_B);
  k_concat_attr<<<G1((long long)Hn * 32), 256, 0, stream>>>(pf(IN_HLL), 4, pf(P_HTR), 8, Hn, XH_B, 32);
  k_concat_attr<<<G1((long long)E_E2H * 32), 256, 0, stream>>>(pf(IN_E2H_A), 3, pf(P_E2HTR), 8, E_E2H, EAIN_B, 32);
  run_mlp(encSrc,  XERA_B, 224, ERA,   nullptr, 0, S_F, S_B, Cc);
  run_mlp(encDst,  XH_B,   32,  Hn,    nullptr, 0, D_F, D_B, Cc);
  run_mlp(encEdge, EAIN_B, 32,  E_E2H, nullptr, 0, EA_F, EA_B, Cc);
  run_block(encBE, encBN, S_F, Cc, D_F, D_B, Cc, EA_F, EA_B, Cc, Cc,
            e2h_src, e2h_dst, E_E2H, Hn, 768, 512);

  // ================= processor (h -> h, noise channel appended) =================
  k_build_xp<<<G1((long long)Hn * PCP), 256, 0, stream>>>(D_F, pf(IN_Z), XP_F, XP_B);
  k_concat_attr<<<G1((long long)E_H2H * 32), 256, 0, stream>>>(pf(IN_H2H_A), 3, pf(P_H2HTR), 8, E_H2H, EAIN_B, 32);
  run_mlp(prEdge, EAIN_B, 32, E_H2H, nullptr, 0, EA_F, EA_B, PCP);
  run_block(prB0E, prB0N, XP_F, PCP, XP_F, XP_B, PCP, EA_F, EA_B, PCP, PCr,
            h2h_src, h2h_dst, E_H2H, Hn, 800, 544);
  run_block(prB1E, prB1N, XP_F, PCP, XP_F, XP_B, PCP, EA_F, EA_B, PCP, PCr,
            h2h_src, h2h_dst, E_H2H, Hn, 800, 544);
  k_residual_slice<<<G1((long long)Hn * Cc), 256, 0, stream>>>(XP_F, D_F, PR_F, PR_B);

  // ================= decoder (h -> era) =================
  run_mlp(decSrc, PR_B, Cc, Hn,  nullptr, 0, DS_F, DS_B, Cc);
  run_mlp(decDst, S_B,  Cc, ERA, nullptr, 0, DD_F, DD_B, Cc);
  k_concat_attr<<<G1((long long)E_H2E * 32), 256, 0, stream>>>(pf(IN_H2E_A), 3, pf(P_H2ETR), 8, E_H2E, EAIN_B, 32);
  run_mlp(decEdge, EAIN_B, 32, E_H2E, nullptr, 0, EA_F, EA_B, Cc);
  run_block(decBE, decBN, DS_F, Cc, DD_F, DD_B, Cc, EA_F, EA_B, Cc, Cc,
            h2e_src, h2e_dst, E_H2E, ERA, 768, 512);

  // final MLP (no LN, layer2 no act) + residual vs x[:,:,-1,:, :98]
  gemmL(DD_B, Cc, FW0, Cc, pf(PD_FIN + 2), Cc, ERA, Cc, Cc, nullptr, 0, HID_B, Cc, 1);
  gemmL(HID_B, Cc, FW1, Cc, pf(PD_FIN + 3), 98, ERA, 128, Cc, GOUT_F, 128, nullptr, 0, 0);
  k_final_out<<<G1((long long)ERA * 98), 256, 0, stream>>>(GOUT_F, x, (float*)d_out);
}
